// GNN_19937238188764
// MI455X (gfx1250) — matrix operand order repaired and checked by reading, not compiled, not run on hardware
//
#include <hip/hip_runtime.h>
#include <hip/hip_bf16.h>

// ---------------------------------------------------------------------------
// GNN-GRU fused kernel for MI455X (gfx1250, wave32, WMMA 16x16x32 f16).
// One workgroup per sentence; entire 10-iteration loop LDS/register-resident.
// Round 4: laundered weight/bias pointers are re-cast to address space 1 so
// B-fragment traffic uses global_load_b128 (saddr + 32-bit voffset) instead of
// flat_load (which couples LOADcnt with DScnt and ties up the LDS path).
// ---------------------------------------------------------------------------

typedef __attribute__((ext_vector_type(16))) _Float16 v16h;
typedef __attribute__((ext_vector_type(8)))  _Float16 v8h;
typedef __attribute__((ext_vector_type(8)))  float    v8f;

#define Hdim 128
#define HH   16384      // 128*128
#define G3H  49152      // 384*128

// Opaque pointer redefinition: loads through p cannot be hoisted above this.
#define LAUNDER(p) asm volatile("" : "+s"(p))

// Workspace layout (in _Float16 elements)
#define WS_GNNWT 0                      // 16384  gnn_w transposed  [n][k]
#define WS_WIH   16384                  // 49152  W_ih  [384][128]  (n-major)
#define WS_WHH   65536                  // 49152  W_hh  [384][128]  (n-major)
#define WS_MW1   114688                 // 16384  mW1   [128][128]  (n-major)
#define WS_MW2   131072                 // 16384  mW2   [128][128]  (n-major)
#define WS_MW3   147456                 // 128    mW3   [128]

// ---- address-space-1 (global) load helpers: saddr + 32-bit offset form ----
__device__ __forceinline__ v16h load16h_g(const _Float16* base, int idx) {
  __attribute__((address_space(1))) const _Float16* g =
      (__attribute__((address_space(1))) const _Float16*)base;
  return *(__attribute__((address_space(1))) const v16h*)(g + idx);
}
__device__ __forceinline__ float loadf_g(const float* base, int idx) {
  __attribute__((address_space(1))) const float* g =
      (__attribute__((address_space(1))) const float*)base;
  return g[idx];
}

// ---- WMMA fragment loaders (CDNA5 ISA §7.12.2 layouts) ----

// A 16x32 f16 fragment from row-major f16 LDS (ld=128).
__device__ __forceinline__ v16h load_a_rm(const _Float16* A, int m0, int k0, int lane) {
  const _Float16* p = A + (m0 + (lane & 15)) * Hdim + k0 + ((lane >> 4) << 3);
  v8h lo = *(const v8h*)(p);
  v8h hi = *(const v8h*)(p + 16);
  return __builtin_shufflevector(lo, hi, 0,1,2,3,4,5,6,7,8,9,10,11,12,13,14,15);
}

// A 16x32 f16 fragment from row-major f32 global (ld=128), converted in-reg.
__device__ __forceinline__ v16h load_a_f32(const float* A, int m0, int k0, int lane) {
  const float* p = A + (m0 + (lane & 15)) * Hdim + k0 + ((lane >> 4) << 3);
  float4 a0 = *(const float4*)(p);
  float4 a1 = *(const float4*)(p + 4);
  float4 b0 = *(const float4*)(p + 16);
  float4 b1 = *(const float4*)(p + 20);
  v16h r;
  r[0]=(_Float16)a0.x;  r[1]=(_Float16)a0.y;  r[2]=(_Float16)a0.z;  r[3]=(_Float16)a0.w;
  r[4]=(_Float16)a1.x;  r[5]=(_Float16)a1.y;  r[6]=(_Float16)a1.z;  r[7]=(_Float16)a1.w;
  r[8]=(_Float16)b0.x;  r[9]=(_Float16)b0.y;  r[10]=(_Float16)b0.z; r[11]=(_Float16)b0.w;
  r[12]=(_Float16)b1.x; r[13]=(_Float16)b1.y; r[14]=(_Float16)b1.z; r[15]=(_Float16)b1.w;
  return r;
}

// B 32x16 f16 fragment from an n-major [N][K] LDS source (ld=128): two 16B chunks (k 8h..8h+7, 16+8h..).
__device__ __forceinline__ v16h load_b_lds(const _Float16* BT, int n0, int k0, int lane) {
  const _Float16* p = BT + (n0 + (lane & 15)) * Hdim + k0 + ((lane >> 4) << 3);
  v8h lo = *(const v8h*)(p);
  v8h hi = *(const v8h*)(p + 16);
  return __builtin_shufflevector(lo, hi, 0,1,2,3,4,5,6,7,8,9,10,11,12,13,14,15);
}

// B 32x16 f16 fragment from an n-major [N][K] GLOBAL source (ld=128).
__device__ __forceinline__ v16h load_b_g(const _Float16* BT, int n0, int k0, int lane) {
  int idx = (n0 + (lane & 15)) * Hdim + k0 + ((lane >> 4) << 3);
  __attribute__((address_space(1))) const _Float16* g =
      (__attribute__((address_space(1))) const _Float16*)BT;
  v8h lo = *(__attribute__((address_space(1))) const v8h*)(g + idx);
  v8h hi = *(__attribute__((address_space(1))) const v8h*)(g + idx + 16);
  return __builtin_shufflevector(lo, hi, 0,1,2,3,4,5,6,7,8,9,10,11,12,13,14,15);
}

__device__ __forceinline__ v8f wmma_f16(v16h a, v16h b, v8f c) {
  return __builtin_amdgcn_wmma_f32_16x16x32_f16(false, a, false, b, (short)0, c,
                                                false, false);
}

__device__ __forceinline__ float fsigm(float x) { return 1.0f / (1.0f + __expf(-x)); }
__device__ __forceinline__ float ftanh(float x) {
  float e = __expf(2.0f * x);
  return 1.0f - 2.0f / (e + 1.0f);
}

// ---------------------------------------------------------------------------
// Prep: convert weights to f16 n-major into workspace.
// ---------------------------------------------------------------------------
__global__ __launch_bounds__(256) void gnn_prep_f16(
    const float* __restrict__ gnn_w, const float* __restrict__ W_ih,
    const float* __restrict__ W_hh,  const float* __restrict__ mW1,
    const float* __restrict__ mW2,   const float* __restrict__ mW3,
    _Float16* __restrict__ ws) {
  int i = blockIdx.x * 256 + threadIdx.x;      // grid covers 49152
  if (i < HH) {
    int k = i >> 7, n = i & 127;               // gnn_w is [k][n]; store [n][k]
    ws[WS_GNNWT + n * Hdim + k] = (_Float16)gnn_w[i];
    ws[WS_MW1 + i] = (_Float16)mW1[i];         // x @ W.T => rows already n-major
    ws[WS_MW2 + i] = (_Float16)mW2[i];
  }
  if (i < G3H) {
    ws[WS_WIH + i] = (_Float16)W_ih[i];        // [384][128] row = output n
    ws[WS_WHH + i] = (_Float16)W_hh[i];
  }
  if (i < Hdim) ws[WS_MW3 + i] = (_Float16)mW3[i];
}

// ---------------------------------------------------------------------------
// Main fused kernel. Dynamic LDS 98816 B:
//   nf16  [128][128] f16  @0       (A operand, own strip per wave)
//   tmpT  [128][128] f16  @32768   (n-major tmp; row-major h1 in MLP)
//   agg16 [128][128] f16  @65536   (row-major agg; row-major h2 in MLP)
//   idxs  [128] int       @98304
// ---------------------------------------------------------------------------
__global__ __launch_bounds__(256) void gnn_fused(
    const int*   __restrict__ input_var,   // [16,128,128]
    const float* __restrict__ adjacency,   // [16,128,128,128]
    const float* __restrict__ emb,         // [50000,128]
    const _Float16* __restrict__ ws,
    const float* __restrict__ gnn_b,       // [128]
    const float* __restrict__ b_ih,        // [384]
    const float* __restrict__ b_hh,        // [384]
    const float* __restrict__ mb1,         // [128]
    const float* __restrict__ mb2,         // [128]
    const float* __restrict__ mb3,         // [1]
    float* __restrict__ out)               // [16, 16384]
{
  extern __shared__ __align__(128) char smem[];
  _Float16* nf16  = (_Float16*)(smem);
  _Float16* tmpT  = (_Float16*)(smem + 32768);
  _Float16* agg16 = (_Float16*)(smem + 65536);
  int*      idxs  = (int*)     (smem + 98304);

  const int sent = blockIdx.x;           // b*128 + s
  const int tid  = threadIdx.x;
  const int lane = tid & 31;
  const int wave = tid >> 5;
  const int m0   = wave << 4;            // this wave's 16-row strip
  const int nlo  = lane & 15;
  const int mb   = m0 + ((lane >> 4) << 3);  // C-fragment row base for this lane

  const _Float16* gnn_wT = ws + WS_GNNWT;
  const _Float16* Wih16  = ws + WS_WIH;
  const _Float16* Whh16  = ws + WS_WHH;
  const _Float16* mW1_16 = ws + WS_MW1;
  const _Float16* mW2_16 = ws + WS_MW2;
  const _Float16* mW3_16 = ws + WS_MW3;
  const float*    gbp    = gnn_b;
  const float*    bip    = b_ih;
  const float*    bhp    = b_hh;

  // Warm WGP$/L2 with GRU weights (gfx1250 global_prefetch_b8): 192KB contiguous.
  {
    const char* wp = (const char*)Wih16;
    for (int i = tid; i < 1536; i += 256) __builtin_prefetch(wp + i * 128, 0, 3);
  }

  if (tid < 128) idxs[tid] = input_var[(size_t)sent * 128 + tid];
  __syncthreads();

  // Embedding gather -> nf16 (coalesced; emb fits in L2).
  for (int o = tid; o < HH; o += 256) {
    int l = o >> 7, h = o & 127;
    nf16[o] = (_Float16)emb[(size_t)idxs[l] * Hdim + h];
  }

  // Adjacency A-fragments: loop-invariant for all 10 iterations -> registers.
  v16h adjF[4];
  {
    const float* abase = adjacency + (size_t)sent * HH;
#pragma unroll
    for (int kk = 0; kk < 4; ++kk) adjF[kk] = load_a_f32(abase, m0, kk * 32, lane);
  }

  // f32 hidden state in registers, C-fragment layout: hstate[nt*8+r] = h[mb+r][nt*16+nlo]
  float hstate[64];
#pragma unroll
  for (int nt = 0; nt < 8; ++nt)
#pragma unroll
    for (int r = 0; r < 8; ++r)
      hstate[nt * 8 + r] = emb[(size_t)idxs[mb + r] * Hdim + nt * 16 + nlo];

  __syncthreads();   // nf16 complete

  // ---- 10 GNN-GRU iterations ----
  for (int it = 0; it < 10; ++it) {
    // Preload nf A-fragments once (own strip); reused by mm1 and the gate matmuls.
    v16h nfF[4];
#pragma unroll
    for (int kk = 0; kk < 4; ++kk) nfF[kk] = load_a_rm(nf16, m0, kk * 32, lane);

    // mm1: tmp = nf @ gnn_w + gnn_b  -> tmpT (n-major f16)
#pragma unroll
    for (int nt = 0; nt < 8; ++nt) {
      LAUNDER(gnn_wT);                 // bound in-flight B-frags; no LICM hoist
      int n0 = nt << 4;
      v8f acc = {};
#pragma unroll
      for (int kk = 0; kk < 4; ++kk)
        acc = wmma_f16(nfF[kk], load_b_g(gnn_wT, n0, kk * 32, lane), acc);
      LAUNDER(gbp);
      float bias = loadf_g(gbp, n0 + nlo);
      v8h pk;
#pragma unroll
      for (int r = 0; r < 8; ++r) pk[r] = (_Float16)(acc[r] + bias);
      *(v8h*)(tmpT + (n0 + nlo) * Hdim + mb) = pk;   // one ds_store_b128
    }
    __syncthreads();   // tmpT complete across waves

    // mm2: agg = adj @ tmp -> agg16 (row-major, own strip only)
#pragma unroll
    for (int nt = 0; nt < 8; ++nt) {
      int n0 = nt << 4;
      v8f acc = {};
#pragma unroll
      for (int kk = 0; kk < 4; ++kk)
        acc = wmma_f16(adjF[kk], load_b_lds(tmpT, n0, kk * 32, lane), acc);
#pragma unroll
      for (int r = 0; r < 8; ++r) agg16[(mb + r) * Hdim + n0 + nlo] = (_Float16)acc[r];
    }

    // agg A-fragments (own strip; same-wave DS ops are in-order).
    v16h agF[4];
#pragma unroll
    for (int kk = 0; kk < 4; ++kk) agF[kk] = load_a_rm(agg16, m0, kk * 32, lane);

    // mm3/mm4 + GRU gates; hstate in registers, nf16 updated in place
    // (safe: nf A-data for this iteration already lives in nfF registers).
#pragma unroll
    for (int nt = 0; nt < 8; ++nt) {
      int n0 = nt << 4;
      v8f ir = {}, iz = {}, in_ = {}, hr = {}, hz = {}, hn = {};
#pragma unroll
      for (int kk = 0; kk < 4; ++kk) {
        LAUNDER(Wih16);                // <=6 weight frags in flight per K-step
        LAUNDER(Whh16);
        int k0 = kk * 32;
        ir  = wmma_f16(agF[kk], load_b_g(Wih16, n0,       k0, lane), ir);
        iz  = wmma_f16(agF[kk], load_b_g(Wih16, 128 + n0, k0, lane), iz);
        in_ = wmma_f16(agF[kk], load_b_g(Wih16, 256 + n0, k0, lane), in_);
        hr  = wmma_f16(nfF[kk], load_b_g(Whh16, n0,       k0, lane), hr);
        hz  = wmma_f16(nfF[kk], load_b_g(Whh16, 128 + n0, k0, lane), hz);
        hn  = wmma_f16(nfF[kk], load_b_g(Whh16, 256 + n0, k0, lane), hn);
      }
      LAUNDER(bip);
      LAUNDER(bhp);
      int n = n0 + nlo;
      float bir = loadf_g(bip, n),       biz = loadf_g(bip, 128 + n);
      float bin = loadf_g(bip, 256 + n), bhr = loadf_g(bhp, n);
      float bhz = loadf_g(bhp, 128 + n), bhn = loadf_g(bhp, 256 + n);
#pragma unroll
      for (int r = 0; r < 8; ++r) {
        float rg = fsigm((ir[r] + bir) + (hr[r] + bhr));
        float z  = fsigm((iz[r] + biz) + (hz[r] + bhz));
        float nn = ftanh((in_[r] + bin) + rg * (hn[r] + bhn));
        float hnew = (1.0f - z) * nn + z * hstate[nt * 8 + r];
        hstate[nt * 8 + r] = hnew;
        nf16[(mb + r) * Hdim + n] = (_Float16)hnew;
      }
    }
    __syncthreads();   // all waves past mm2's tmpT reads; nf16/gates complete
  }

  // ---- mention predictor MLP ----
  v16h xF[4];
#pragma unroll
  for (int kk = 0; kk < 4; ++kk) xF[kk] = load_a_rm(nf16, m0, kk * 32, lane);
#pragma unroll
  for (int nt = 0; nt < 8; ++nt) {          // h1 = tanh(x @ mW1^T + mb1) -> tmpT
    LAUNDER(mW1_16);
    int n0 = nt << 4;
    v8f acc = {};
#pragma unroll
    for (int kk = 0; kk < 4; ++kk)
      acc = wmma_f16(xF[kk], load_b_g(mW1_16, n0, kk * 32, lane), acc);
    float bias = loadf_g(mb1, n0 + nlo);
#pragma unroll
    for (int r = 0; r < 8; ++r)
      tmpT[(mb + r) * Hdim + n0 + nlo] = (_Float16)ftanh(acc[r] + bias);
  }
  v16h h1F[4];
#pragma unroll
  for (int kk = 0; kk < 4; ++kk) h1F[kk] = load_a_rm(tmpT, m0, kk * 32, lane);
#pragma unroll
  for (int nt = 0; nt < 8; ++nt) {          // h2 = tanh(h1 @ mW2^T + mb2) -> agg16
    LAUNDER(mW2_16);
    int n0 = nt << 4;
    v8f acc = {};
#pragma unroll
    for (int kk = 0; kk < 4; ++kk)
      acc = wmma_f16(h1F[kk], load_b_g(mW2_16, n0, kk * 32, lane), acc);
    float bias = loadf_g(mb2, n0 + nlo);
#pragma unroll
    for (int r = 0; r < 8; ++r)
      agg16[(mb + r) * Hdim + n0 + nlo] = (_Float16)ftanh(acc[r] + bias);
  }
  __syncthreads();   // final reduce crosses wave strips

  // score[l] = tanh(dot(h2[l], mW3) + mb3)
  if (tid < 128) {
    float sum = loadf_g(mb3, 0);
#pragma unroll 8
    for (int k = 0; k < Hdim; ++k)
      sum += (float)agg16[tid * Hdim + k] * (float)mW3_16[k];
    out[(size_t)sent * 128 + tid] = ftanh(sum);
  }
}

// ---------------------------------------------------------------------------
extern "C" void kernel_launch(void* const* d_in, const int* in_sizes, int n_in,
                              void* d_out, int out_size, void* d_ws, size_t ws_size,
                              hipStream_t stream) {
  const int*   input_var = (const int*)  d_in[0];
  const float* adjacency = (const float*)d_in[1];
  const float* emb       = (const float*)d_in[2];
  const float* gnn_w     = (const float*)d_in[3];
  const float* gnn_b     = (const float*)d_in[4];
  const float* W_ih      = (const float*)d_in[5];
  const float* W_hh      = (const float*)d_in[6];
  const float* b_ih      = (const float*)d_in[7];
  const float* b_hh      = (const float*)d_in[8];
  const float* mW1       = (const float*)d_in[9];
  const float* mb1       = (const float*)d_in[10];
  const float* mW2       = (const float*)d_in[11];
  const float* mb2       = (const float*)d_in[12];
  const float* mW3       = (const float*)d_in[13];
  const float* mb3       = (const float*)d_in[14];
  _Float16*    ws16      = (_Float16*)d_ws;       // needs ~289 KB
  float*       outp      = (float*)d_out;

  gnn_prep_f16<<<192, 256, 0, stream>>>(gnn_w, W_ih, W_hh, mW1, mW2, mW3, ws16);

  // 2048 sentences, 256 threads (8 waves), 98816 B dynamic LDS (3 WGs/WGP).
  gnn_fused<<<2048, 256, 98816, stream>>>(input_var, adjacency, emb, ws16,
                                          gnn_b, b_ih, b_hh, mb1, mb2, mb3, outp);
}
